// FeatureMLP_68083821576979
// MI455X (gfx1250) — compile-verified, matrix operand order checked by reading, not run on hardware
//
#include <hip/hip_runtime.h>
#include <hip/hip_bf16.h>
#include <math.h>

// ---------------------------------------------------------------------------
// FeatureMLP for MI455X (gfx1250, wave32, WMMA).
//   K0 (k_twiddle): writes DFT cos/sin twiddles as f16 WMMA B-fragments
//       (fragment-order layout, 144 KB, L2-resident for all K1 blocks).
//   K1 (k_features16): 16 (b,c) rows per block; time stats streamed from
//       global; 256-pt windowed DFT as WMMA GEMM with vectorized fragment
//       loads: per N-tile just 8 x (2 vector loads + 2 v_wmma).
//   K2: channel aggregation -> 35 features per batch row (padded to 64)
//   K3: MLP 35->128->64->32->1 with v_wmma_f32_16x16x32_f16, one wave/block
// Input order (setup_inputs dict flattening):
//   0:x 1:W1 2:b1 3:W2 4:b2 5:W3 6:b3 7:W4 8:b4
//   9:g1 10:be1 11:rm1 12:rv1 13:g2 14:be2 15:rm2 16:rv2 17:g3 18:be3 19:rm3 20:rv3
// ---------------------------------------------------------------------------

typedef __attribute__((ext_vector_type(16))) _Float16 v16h;
typedef __attribute__((ext_vector_type(8)))  float    v8f;

union AFrag { v16h v; _Float16 e[16]; };
union CFrag { v8f  v; float    e[8];  };

#define T_LEN 1000
#define NSEG  256
#define NBIN  129
#define NCH   129
#define NBATCH 512
#define DF    0.390625f          // 100/256
#define FEAT_STRIDE 64           // features padded 35 -> 64 for WMMA K-tiles
#define ROWS_PER_BLK 16
#define PSDS 144                 // padded bin stride (9 N-tiles of 16)
#define NTILES 9
#define KTILES 8
#define TW_FRAGS (2 * NTILES * KTILES * 32)     // v16h fragments (cos+sin)
#define TW_ELEMS (TW_FRAGS * 16)                // 73728 f16

__device__ __forceinline__ int kmap(int i, int g) {
    // CDNA5 16-bit A/B fragment K index for element i, lane-half g (ISA 7.12.2)
    return (i & 7) + 8 * g + 16 * (i >> 3);
}

// ---- reduction within groups of 16 threads (blockDim.x == 256) ------------
__device__ __forceinline__ float groupReduceSum16(float v, float* s, int tid) {
    __syncthreads();
    s[tid] = v;
    __syncthreads();
    if ((tid & 15) < 8) s[tid] += s[tid + 8];
    __syncthreads();
    if ((tid & 15) < 4) s[tid] += s[tid + 4];
    __syncthreads();
    if ((tid & 15) < 2) s[tid] += s[tid + 2];
    __syncthreads();
    if ((tid & 15) < 1) s[tid] += s[tid + 1];
    __syncthreads();
    return s[tid & ~15];
}

// ---- full block reductions (blockDim.x == 256) for K2 ---------------------
__device__ __forceinline__ float blockReduceSum(float v, float* s, int tid) {
    __syncthreads();
    s[tid] = v;
    __syncthreads();
    for (int off = 128; off > 0; off >>= 1) {
        if (tid < off) s[tid] += s[tid + off];
        __syncthreads();
    }
    return s[0];
}

__device__ __forceinline__ float blockReduceMax(float v, float* s, int tid) {
    __syncthreads();
    s[tid] = v;
    __syncthreads();
    for (int off = 128; off > 0; off >>= 1) {
        if (tid < off) s[tid] = fmaxf(s[tid], s[tid + off]);
        __syncthreads();
    }
    return s[0];
}

// ---------------------------------------------------------------------------
// K0: twiddle fragments.  Layout: v16h index = ((p*NTILES+nt)*KTILES+kt)*32+lane
//     element i of lane -> B[k][nb], k = kt*32+kmap(i,g), nb = nt*16+(lane&15)
//     p=0: cos(2*pi*k*nb/256), p=1: sin (via cos phase offset, sign irrelevant
//     because only re^2+im^2 is consumed).
// ---------------------------------------------------------------------------
__global__ __launch_bounds__(256) void k_twiddle(_Float16* __restrict__ tw) {
    const int idx = blockIdx.x * 256 + threadIdx.x;
    if (idx >= TW_ELEMS) return;
    const int i    = idx & 15;
    const int lane = (idx >> 4) & 31;
    const int kt   = (idx >> 9) & 7;
    const int rest = idx >> 12;
    const int nt   = rest % NTILES;
    const int p    = rest / NTILES;
    const int mloc = lane & 15;
    const int g    = lane >> 4;
    const int nb   = nt * 16 + mloc;
    const int k    = kt * 32 + kmap(i, g);
    int j = (k * nb) & 255;
    if (p) j = (j + 192) & 255;          // sin(t) = cos(t - pi/2)
    tw[idx] = (_Float16)cosf(6.28318530718f * (float)j * (1.0f / 256.0f));
}

// ---------------------------------------------------------------------------
// K1: one block per 16 (b,c) rows. 256 threads = 8 wave32.
// perch[row*16 + j]: 0..4 band powers, 5:ent, 6:pf, 7:sc, 8:chm, 9:chs,
//                    10:var_x, 11:mob, 12:comp, 13:zc
// ---------------------------------------------------------------------------
__global__ __launch_bounds__(256) void k_features16(const float* __restrict__ x,
                                                    const _Float16* __restrict__ tw,
                                                    float* __restrict__ perch) {
    __shared__ float s_x256[ROWS_PER_BLK * NSEG];   // first 256 samples/row
    __shared__ float s_psd[ROWS_PER_BLK * PSDS];
    __shared__ float s_win[NSEG];                   // Hann window
    __shared__ float s_red[256];

    const int tid = threadIdx.x;
    const size_t row0 = (size_t)blockIdx.x * ROWS_PER_BLK;

    // stage first 256 samples of each row (coalesced) + window table
    for (int it = 0; it < ROWS_PER_BLK; ++it)
        s_x256[it * NSEG + tid] = x[(row0 + it) * T_LEN + tid];
    s_win[tid] = 0.5f * (1.0f - cosf(6.28318530718f * (float)tid * (1.0f / 256.0f)));

    // ---- time-domain stats: 16 threads per channel, streamed from global ----
    const int ch = tid >> 4;
    const int sub = tid & 15;
    const float* row = x + (row0 + ch) * T_LEN;
    __builtin_prefetch(row, 0, 3);     // global_prefetch_b8
    float sx = 0.f, sxx = 0.f, sd1 = 0.f, sd2 = 0.f, lzc = 0.f;
    for (int i = sub; i < T_LEN; i += 16) {
        const float a = row[i];
        sx += a;
        sxx += a * a;
        if (i < T_LEN - 1) {
            const float b = row[i + 1];
            const float d1 = b - a;
            sd1 += d1 * d1;
            const float sa = (a > 0.f) ? 1.f : ((a < 0.f) ? -1.f : 0.f);
            const float sb = (b > 0.f) ? 1.f : ((b < 0.f) ? -1.f : 0.f);
            lzc += (sa != sb) ? 1.f : 0.f;
            if (i < T_LEN - 2) {
                const float d2 = (row[i + 2] - b) - d1;
                sd2 += d2 * d2;
            }
        }
    }
    sx  = groupReduceSum16(sx,  s_red, tid);
    sxx = groupReduceSum16(sxx, s_red, tid);
    sd1 = groupReduceSum16(sd1, s_red, tid);
    sd2 = groupReduceSum16(sd2, s_red, tid);
    lzc = groupReduceSum16(lzc, s_red, tid);
    const float mean  = sx * (1.0f / T_LEN);
    const float var_x = fmaxf((sxx - (float)T_LEN * mean * mean) * (1.0f / 999.0f), 0.f);
    const float md1   = (row[999] - row[0]) * (1.0f / 999.0f);
    const float var_d1 = fmaxf((sd1 - 999.0f * md1 * md1) * (1.0f / 998.0f), 0.f);
    const float md2   = ((row[999] - row[998]) - (row[1] - row[0])) * (1.0f / 998.0f);
    const float var_d2 = fmaxf((sd2 - 998.0f * md2 * md2) * (1.0f / 997.0f), 0.f);
    const float zc = lzc * (1.0f / T_LEN);
    __syncthreads();   // s_x256 and s_win ready

    // ---- DFT as WMMA GEMM: D = Seg(16x256) x Twiddle(256x144), cos & sin ----
    {
        const int lane = tid & 31;
        const int wave = tid >> 5;
        const int mloc = lane & 15;          // A row / B,D column-in-tile
        const int g = lane >> 4;

        // A fragments: per lane two contiguous 8-float runs per k-tile ->
        // vectorized ds_load_b128 of segment and window, fused mul + cvt.
        AFrag a[KTILES];
        #pragma unroll
        for (int kt = 0; kt < KTILES; ++kt) {
            const int base = kt * 32 + 8 * g;
            const float4* xa = (const float4*)(s_x256 + mloc * NSEG + base);
            const float4* wv = (const float4*)(s_win + base);
            const float4 x0 = xa[0], x1 = xa[1], x2 = xa[4], x3 = xa[5];
            const float4 w0 = wv[0], w1 = wv[1], w2 = wv[4], w3 = wv[5];
            a[kt].e[0]  = (_Float16)(x0.x * w0.x);
            a[kt].e[1]  = (_Float16)(x0.y * w0.y);
            a[kt].e[2]  = (_Float16)(x0.z * w0.z);
            a[kt].e[3]  = (_Float16)(x0.w * w0.w);
            a[kt].e[4]  = (_Float16)(x1.x * w1.x);
            a[kt].e[5]  = (_Float16)(x1.y * w1.y);
            a[kt].e[6]  = (_Float16)(x1.z * w1.z);
            a[kt].e[7]  = (_Float16)(x1.w * w1.w);
            a[kt].e[8]  = (_Float16)(x2.x * w2.x);
            a[kt].e[9]  = (_Float16)(x2.y * w2.y);
            a[kt].e[10] = (_Float16)(x2.z * w2.z);
            a[kt].e[11] = (_Float16)(x2.w * w2.w);
            a[kt].e[12] = (_Float16)(x3.x * w3.x);
            a[kt].e[13] = (_Float16)(x3.y * w3.y);
            a[kt].e[14] = (_Float16)(x3.z * w3.z);
            a[kt].e[15] = (_Float16)(x3.w * w3.w);
        }

        const v16h* twv = (const v16h*)tw;   // fragment-order twiddles (L2-hot)
        for (int nt = wave; nt < NTILES; nt += 8) {
            const int nb = nt * 16 + mloc;   // frequency bin column
            CFrag re, im;
            #pragma unroll
            for (int r = 0; r < 8; ++r) { re.e[r] = 0.f; im.e[r] = 0.f; }
            #pragma unroll
            for (int kt = 0; kt < KTILES; ++kt) {
                const v16h bc = twv[(nt * KTILES + kt) * 32 + lane];
                const v16h bs = twv[((NTILES + nt) * KTILES + kt) * 32 + lane];
                re.v = __builtin_amdgcn_wmma_f32_16x16x32_f16(false, a[kt].v, false, bc,
                                                              (short)0, re.v, false, false);
                im.v = __builtin_amdgcn_wmma_f32_16x16x32_f16(false, a[kt].v, false, bs,
                                                              (short)0, im.v, false, false);
            }
            #pragma unroll
            for (int r = 0; r < 8; ++r) {
                const int M = r + 8 * g;     // channel within block
                s_psd[M * PSDS + nb] =
                    (re.e[r] * re.e[r] + im.e[r] * im.e[r]) * (1.0f / NSEG);
            }
        }
    }
    __syncthreads();   // s_psd ready (bins >= 129 are padding, never read)

    // ---- spectral scalars: 16 threads per channel ----
    const float* psd = s_psd + ch * PSDS;
    float lp = 0.f;
    for (int n = sub; n < NBIN; n += 16) lp += psd[n];
    const float psum = groupReduceSum16(lp, s_red, tid);
    float lent = 0.f, lsc = 0.f;
    for (int n = sub; n < NBIN; n += 16) {
        const float pn = (psum > 0.f) ? psd[n] / psum : psd[n] + 1e-10f;
        lent += -pn * log2f(pn + 1e-10f);
        lsc  += ((float)n * DF) * pn;
    }
    const float ent = groupReduceSum16(lent, s_red, tid);
    const float sc  = groupReduceSum16(lsc,  s_red, tid);

    float* out = perch + (row0 + ch) * 16;
    if (sub < 5) {
        const int A[5]   = {2, 11, 21, 34, 77};
        const int Bnd[5] = {11, 21, 34, 77, 129};
        const int a = A[sub], b = Bnd[sub];
        float ssum = 0.f;
        for (int k = a; k < b; ++k) ssum += psd[k];
        out[sub] = DF * (ssum - 0.5f * (psd[a] + psd[b - 1]));
    }
    if (sub == 5) {
        float best = psd[0];
        int bi = 0;
        for (int k = 1; k < NBIN; ++k)
            if (psd[k] > best) { best = psd[k]; bi = k; }
        out[6] = (float)bi * DF;
    }
    if (sub == 0) {
        out[5] = ent;
        out[7] = sc;
        out[8] = mean;
        out[9] = sqrtf(var_x);
        out[10] = var_x;
        const float mob  = (var_x > 0.f) ? sqrtf(var_d1 / var_x) : 0.f;
        const float comp = (mob > 0.f && var_d1 > 0.f)
                             ? sqrtf(var_d2 / var_d1) / mob : 0.f;
        out[11] = mob;
        out[12] = comp;
        out[13] = zc;
    }
}

// ---------------------------------------------------------------------------
// K2: one block per batch row; reduce 129 channels -> 35 features (+pad to 64)
// ---------------------------------------------------------------------------
__global__ __launch_bounds__(256) void k_aggregate(const float* __restrict__ perch,
                                                   float* __restrict__ feats) {
    __shared__ float s_red[256];
    const int tid = threadIdx.x;
    const int b = blockIdx.x;
    const bool act = (tid < NCH);
    const int ch = act ? tid : 0;
    const float* src = perch + ((size_t)b * NCH + ch) * 16;
    float v[14];
    for (int j = 0; j < 14; ++j) v[j] = act ? src[j] : 0.f;

    const float invC = 1.0f / 129.0f;
    const float invC1 = 1.0f / 128.0f;
    float feat[35];
    float m[5];
    for (int i = 0; i < 5; ++i) {
        const float mu = blockReduceSum(act ? v[i] : 0.f, s_red, tid) * invC;
        const float d = act ? (v[i] - mu) : 0.f;
        const float sd = sqrtf(blockReduceSum(d * d, s_red, tid) * invC1);
        const float mx = blockReduceMax(act ? v[i] : -3.4e38f, s_red, tid);
        feat[3 * i] = mu; feat[3 * i + 1] = sd; feat[3 * i + 2] = mx;
        m[i] = mu;
    }
    const float total = m[0] + m[1] + m[2] + m[3] + m[4];
    for (int i = 0; i < 5; ++i)
        feat[15 + i] = (total > 0.f) ? m[i] / total : 0.f;

    int fi = 20;
    for (int j = 5; j <= 7; ++j) {       // ent, pf, sc: mean+std
        const float mu = blockReduceSum(act ? v[j] : 0.f, s_red, tid) * invC;
        const float d = act ? (v[j] - mu) : 0.f;
        const float sd = sqrtf(blockReduceSum(d * d, s_red, tid) * invC1);
        feat[fi++] = mu; feat[fi++] = sd;
    }
    for (int j = 8; j <= 9; ++j) {       // chm, chs: mean+std
        const float mu = blockReduceSum(act ? v[j] : 0.f, s_red, tid) * invC;
        const float d = act ? (v[j] - mu) : 0.f;
        const float sd = sqrtf(blockReduceSum(d * d, s_red, tid) * invC1);
        feat[fi++] = mu; feat[fi++] = sd;
    }
    for (int j = 10; j <= 12; ++j)       // var_x, mob, comp: mean only
        feat[fi++] = blockReduceSum(act ? v[j] : 0.f, s_red, tid) * invC;
    {                                    // zc: mean+std
        const float mu = blockReduceSum(act ? v[13] : 0.f, s_red, tid) * invC;
        const float d = act ? (v[13] - mu) : 0.f;
        const float sd = sqrtf(blockReduceSum(d * d, s_red, tid) * invC1);
        feat[fi++] = mu; feat[fi++] = sd;
    }
    if (tid == 0)
        for (int f = 0; f < 35; ++f) feats[(size_t)b * FEAT_STRIDE + f] = feat[f];
    if (tid >= 35 && tid < FEAT_STRIDE)
        feats[(size_t)b * FEAT_STRIDE + tid] = 0.f;
}

// ---------------------------------------------------------------------------
// K3: WMMA MLP. 32 blocks x 32 threads (one wave32 per block; EXEC all ones).
// ---------------------------------------------------------------------------
__global__ __launch_bounds__(32) void k_mlp(
    const float* __restrict__ feats,
    const float* __restrict__ W1, const float* __restrict__ b1,
    const float* __restrict__ W2, const float* __restrict__ b2,
    const float* __restrict__ W3, const float* __restrict__ b3,
    const float* __restrict__ W4, const float* __restrict__ b4,
    const float* __restrict__ g1, const float* __restrict__ be1,
    const float* __restrict__ rm1, const float* __restrict__ rv1,
    const float* __restrict__ g2, const float* __restrict__ be2,
    const float* __restrict__ rm2, const float* __restrict__ rv2,
    const float* __restrict__ g3, const float* __restrict__ be3,
    const float* __restrict__ rm3, const float* __restrict__ rv3,
    float* __restrict__ out) {
    __shared__ float s_h1[16 * 128];
    __shared__ float s_h2[16 * 64];
    __shared__ float s_h3[16 * 32];

    const int lane = threadIdx.x;
    const int mloc = lane & 15;
    const int g = lane >> 4;
    const int m0 = blockIdx.x * 16;

    AFrag a0, a1;
    #pragma unroll
    for (int i = 0; i < 16; ++i) {
        const int k = kmap(i, g);
        a0.e[i] = (_Float16)feats[(size_t)(m0 + mloc) * FEAT_STRIDE + k];
        a1.e[i] = (_Float16)feats[(size_t)(m0 + mloc) * FEAT_STRIDE + 32 + k];
    }

    // ---- layer 1: 35 -> 128 ----
    for (int nt = 0; nt < 8; ++nt) {
        const int n = nt * 16 + mloc;
        AFrag bf0, bf1;
        #pragma unroll
        for (int i = 0; i < 16; ++i) {
            const int k = kmap(i, g);
            bf0.e[i] = (k < 35) ? (_Float16)W1[k * 128 + n] : (_Float16)0.f;
            const int k1 = 32 + k;
            bf1.e[i] = (k1 < 35) ? (_Float16)W1[k1 * 128 + n] : (_Float16)0.f;
        }
        CFrag acc;
        #pragma unroll
        for (int r = 0; r < 8; ++r) acc.e[r] = 0.f;
        acc.v = __builtin_amdgcn_wmma_f32_16x16x32_f16(false, a0.v, false, bf0.v,
                                                       (short)0, acc.v, false, false);
        acc.v = __builtin_amdgcn_wmma_f32_16x16x32_f16(false, a1.v, false, bf1.v,
                                                       (short)0, acc.v, false, false);
        const float bb = b1[n], mu = rm1[n], iv = rsqrtf(rv1[n] + 1e-5f);
        const float gg = g1[n], bee = be1[n];
        #pragma unroll
        for (int r = 0; r < 8; ++r) {
            const int M = r + 8 * g;
            float h = acc.e[r] + bb;
            h = (h - mu) * iv * gg + bee;
            s_h1[M * 128 + n] = fmaxf(h, 0.f);
        }
    }
    __syncthreads();

    // ---- layer 2: 128 -> 64 ----
    AFrag a2[4];
    #pragma unroll
    for (int kt = 0; kt < 4; ++kt)
        #pragma unroll
        for (int i = 0; i < 16; ++i)
            a2[kt].e[i] = (_Float16)s_h1[mloc * 128 + kt * 32 + kmap(i, g)];
    __syncthreads();
    for (int nt = 0; nt < 4; ++nt) {
        const int n = nt * 16 + mloc;
        CFrag acc;
        #pragma unroll
        for (int r = 0; r < 8; ++r) acc.e[r] = 0.f;
        for (int kt = 0; kt < 4; ++kt) {
            AFrag bf;
            #pragma unroll
            for (int i = 0; i < 16; ++i)
                bf.e[i] = (_Float16)W2[(kt * 32 + kmap(i, g)) * 64 + n];
            acc.v = __builtin_amdgcn_wmma_f32_16x16x32_f16(false, a2[kt].v, false, bf.v,
                                                           (short)0, acc.v, false, false);
        }
        const float bb = b2[n], mu = rm2[n], iv = rsqrtf(rv2[n] + 1e-5f);
        const float gg = g2[n], bee = be2[n];
        #pragma unroll
        for (int r = 0; r < 8; ++r) {
            const int M = r + 8 * g;
            float h = acc.e[r] + bb;
            h = (h - mu) * iv * gg + bee;
            s_h2[M * 64 + n] = fmaxf(h, 0.f);
        }
    }
    __syncthreads();

    // ---- layer 3: 64 -> 32 ----
    AFrag a3[2];
    #pragma unroll
    for (int kt = 0; kt < 2; ++kt)
        #pragma unroll
        for (int i = 0; i < 16; ++i)
            a3[kt].e[i] = (_Float16)s_h2[mloc * 64 + kt * 32 + kmap(i, g)];
    __syncthreads();
    for (int nt = 0; nt < 2; ++nt) {
        const int n = nt * 16 + mloc;
        CFrag acc;
        #pragma unroll
        for (int r = 0; r < 8; ++r) acc.e[r] = 0.f;
        for (int kt = 0; kt < 2; ++kt) {
            AFrag bf;
            #pragma unroll
            for (int i = 0; i < 16; ++i)
                bf.e[i] = (_Float16)W3[(kt * 32 + kmap(i, g)) * 32 + n];
            acc.v = __builtin_amdgcn_wmma_f32_16x16x32_f16(false, a3[kt].v, false, bf.v,
                                                           (short)0, acc.v, false, false);
        }
        const float bb = b3[n], mu = rm3[n], iv = rsqrtf(rv3[n] + 1e-5f);
        const float gg = g3[n], bee = be3[n];
        #pragma unroll
        for (int r = 0; r < 8; ++r) {
            const int M = r + 8 * g;
            float h = acc.e[r] + bb;
            h = (h - mu) * iv * gg + bee;
            s_h3[M * 32 + n] = fmaxf(h, 0.f);
        }
    }
    __syncthreads();

    // ---- layer 4: 32 -> 1, sigmoid, scale to [0.5, 1.5] ----
    if (lane < 16) {
        float acc4 = b4[0];
        #pragma unroll
        for (int k = 0; k < 32; ++k) acc4 += s_h3[lane * 32 + k] * W4[k];
        const float sg = 1.0f / (1.0f + expf(-acc4));
        out[m0 + lane] = 0.5f + sg;
    }
}

// ---------------------------------------------------------------------------
extern "C" void kernel_launch(void* const* d_in, const int* in_sizes, int n_in,
                              void* d_out, int out_size, void* d_ws, size_t ws_size,
                              hipStream_t stream) {
    const float* x  = (const float*)d_in[0];
    const float* W1 = (const float*)d_in[1];  const float* b1 = (const float*)d_in[2];
    const float* W2 = (const float*)d_in[3];  const float* b2 = (const float*)d_in[4];
    const float* W3 = (const float*)d_in[5];  const float* b3 = (const float*)d_in[6];
    const float* W4 = (const float*)d_in[7];  const float* b4 = (const float*)d_in[8];
    const float* g1 = (const float*)d_in[9];  const float* be1 = (const float*)d_in[10];
    const float* rm1 = (const float*)d_in[11]; const float* rv1 = (const float*)d_in[12];
    const float* g2 = (const float*)d_in[13]; const float* be2 = (const float*)d_in[14];
    const float* rm2 = (const float*)d_in[15]; const float* rv2 = (const float*)d_in[16];
    const float* g3 = (const float*)d_in[17]; const float* be3 = (const float*)d_in[18];
    const float* rm3 = (const float*)d_in[19]; const float* rv3 = (const float*)d_in[20];

    float* perch = (float*)d_ws;                              // 512*129*16 f32
    float* feats = perch + (size_t)NBATCH * NCH * 16;         // 512*64 f32
    _Float16* twid = (_Float16*)(feats + (size_t)NBATCH * FEAT_STRIDE);  // 73728 f16

    k_twiddle<<<(TW_ELEMS + 255) / 256, 256, 0, stream>>>(twid);
    k_features16<<<(NBATCH * NCH) / ROWS_PER_BLK, 256, 0, stream>>>(x, twid, perch);
    k_aggregate<<<NBATCH, 256, 0, stream>>>(perch, feats);
    k_mlp<<<NBATCH / 16, 32, 0, stream>>>(feats, W1, b1, W2, b2, W3, b3, W4, b4,
                                          g1, be1, rm1, rv1, g2, be2, rm2, rv2,
                                          g3, be3, rm3, rv3, (float*)d_out);
}
